// GravitationLayer_31147102830703
// MI455X (gfx1250) — matrix operand order repaired and checked by reading, not compiled, not run on hardware
//
#include <hip/hip_runtime.h>
#include <math.h>

// ---------------- problem constants ----------------
#define DM   1024            // d_model
#define NH   16              // heads
#define DH   64              // head dim
#define BATCH 2
#define SEQ  2048
#define ROWS (BATCH*SEQ)     // 4096
#define BH   (BATCH*NH)      // 32
#define EPSV 1e-8f

typedef __attribute__((ext_vector_type(16))) __bf16 v16bf;
typedef __attribute__((ext_vector_type(8)))  __bf16 v8bf;
typedef __attribute__((ext_vector_type(8)))  float  v8f;

// ---------------- WMMA helpers ----------------
__device__ __forceinline__ v8f wmma_bf16(v16bf a, v16bf b, v8f c) {
    // D = A(16x32 bf16) * B(32x16 bf16) + C(16x16 f32)
    return __builtin_amdgcn_wmma_f32_16x16x32_bf16(
        /*neg_a=*/false, a, /*neg_b=*/false, b,
        /*c_mod=*/(short)0, c, /*reuse_a=*/false, /*reuse_b=*/false);
}

// A-operand loader: 16x32 bf16 tile, lane = row (mod 16).
// rowbase points at element (row, k0) of a row-major matrix.
// Layout: elems 0..15 -> K = (lane>>4)*16 + e (contiguous, matches B operands)
__device__ __forceinline__ v16bf load_a16x32(const __bf16* rowbase, int lane) {
    int g = (lane >> 4) << 3;
    v8bf lo = *(const v8bf*)(rowbase + 2 * g);
    v8bf hi = *(const v8bf*)(rowbase + 2 * g + 8);
    return __builtin_shufflevector(lo, hi, 0,1,2,3,4,5,6,7,8,9,10,11,12,13,14,15);
}

// ---------------- fp32 -> bf16 cast ----------------
__global__ void cast_f32_bf16_k(const float* __restrict__ src, __bf16* __restrict__ dst, int n) {
    int i = blockIdx.x * blockDim.x + threadIdx.x;
    int stride = gridDim.x * blockDim.x;
    for (; i < n; i += stride) dst[i] = (__bf16)src[i];
}

// ---------------- bf16 GEMM:  Y[m,n] = sum_k A[m,k]*W[n,k] + bias[n] ----------------
// Per-wave tile: 32x64 (2 M-subtiles x 4 N-subtiles) => 8 WMMA per k-step of 32,
// fed by 2 A-operands + 4 B-operands (~1.5 mem-ops per WMMA instead of 6).
// mode 0: write bf16 to [B,H,S,Dh]        (Q, K)
// mode 1: write bf16 to [B,H,Dh,S]        (V transposed for B-operand streaming)
// mode 2: write f32  to [M,N] = d_out     (final projection)
__global__ void gemm_bf16_k(const __bf16* __restrict__ A, const __bf16* __restrict__ W,
                            const float* __restrict__ bias, int mode,
                            __bf16* __restrict__ out_bf, float* __restrict__ out_f) {
    int tid  = blockIdx.x * blockDim.x + threadIdx.x;
    int wave = tid >> 5;
    int lane = tid & 31;
    const int tilesN = DM / 64;                 // 16
    int tm = (wave / tilesN) * 32;
    int tn = (wave % tilesN) * 64;
    if (tm >= ROWS) return;

    const __bf16* ar0 = A + (size_t)(tm + (lane & 15)) * DM;
    const __bf16* ar1 = ar0 + (size_t)16 * DM;
    const __bf16* br[4];
    #pragma unroll
    for (int ni = 0; ni < 4; ++ni)
        br[ni] = W + (size_t)(tn + ni * 16 + (lane & 15)) * DM + ((lane >> 4) << 4);

    v8f acc[2][4] = {{{}, {}, {}, {}}, {{}, {}, {}, {}}};

    #pragma unroll 2
    for (int k0 = 0; k0 < DM; k0 += 32) {
        // WGP-scope prefetch (locality 3 -> nearest cache): operands are already
        // GL2-resident (192MB L2 holds the whole workspace); we want them in WGP$.
        __builtin_prefetch(ar0 + k0 + 256, 0, 3);   // global_prefetch_b8
        __builtin_prefetch(ar1 + k0 + 256, 0, 3);
        v16bf a0 = load_a16x32(ar0 + k0, lane);
        v16bf a1 = load_a16x32(ar1 + k0, lane);
        #pragma unroll
        for (int ni = 0; ni < 4; ++ni) {
            v16bf b = *(const v16bf*)(br[ni] + k0);
            acc[0][ni] = wmma_bf16(a0, b, acc[0][ni]);
            acc[1][ni] = wmma_bf16(a1, b, acc[1][ni]);
        }
    }

    #pragma unroll
    for (int ni = 0; ni < 4; ++ni) {
        int n  = tn + ni * 16 + (lane & 15);
        float bn = bias[n];
        #pragma unroll
        for (int mi = 0; mi < 2; ++mi) {
            #pragma unroll
            for (int r = 0; r < 8; ++r) {
                int m = tm + mi * 16 + r + ((lane >> 4) << 3);
                float y = acc[mi][ni][r] + bn;
                if (mode == 2) {
                    out_f[(size_t)m * DM + n] = y;
                } else {
                    int b_ = m >> 11, s = m & (SEQ - 1);
                    int h  = n >> 6,  dh = n & (DH - 1);
                    if (mode == 0)
                        out_bf[(((size_t)(b_*NH + h) * SEQ) + s) * DH + dh] = (__bf16)y;
                    else
                        out_bf[(((size_t)(b_*NH + h) * DH) + dh) * SEQ + s] = (__bf16)y;
                }
            }
        }
    }
}

// ---------------- per-row norms: q2, k2, mass = sqrt(k2) over [B*H*S] ----------------
__global__ void norms_k(const __bf16* __restrict__ Qb, const __bf16* __restrict__ Kb,
                        float* __restrict__ q2, float* __restrict__ k2, float* __restrict__ mass) {
    int i = blockIdx.x * blockDim.x + threadIdx.x;
    if (i >= BH * SEQ) return;
    const __bf16* q = Qb + (size_t)i * DH;
    const __bf16* k = Kb + (size_t)i * DH;
    float sq = 0.f, sk = 0.f;
    #pragma unroll 8
    for (int d = 0; d < DH; ++d) {
        float a = (float)q[d]; sq += a * a;
        float b = (float)k[d]; sk += b * b;
    }
    q2[i] = sq; k2[i] = sk; mass[i] = sqrtf(sk);
}

// ---------------- streaming gravitational attention ----------------
// grid = (SEQ/128, B*H), block = 256 (8 waves). Each wave owns 16 query rows.
// Per j-chunk of 32: S = Q Kt via WMMA; w = m_j / (sqrt(max(d2,0))+eps)^2 (VALU);
// stage w in LDS (C-layout -> A-layout transpose); num += W*V via WMMA; den += rowsum(W).
// (G and m_i cancel under row-normalization: w_ij = m_j / d_ij^2.)
__global__ void attn_k(const __bf16* __restrict__ Qb, const __bf16* __restrict__ Kb,
                       const __bf16* __restrict__ Vt,
                       const float* __restrict__ q2, const float* __restrict__ k2,
                       const float* __restrict__ mass,
                       __bf16* __restrict__ AO) {
    __shared__ __align__(32) __bf16 sw[8][16 * 32];   // per-wave 16x32 score staging

    int lane = threadIdx.x & 31;
    int wave = threadIdx.x >> 5;
    int bh   = blockIdx.y;
    int b_   = bh >> 4, h = bh & 15;
    int qbase = blockIdx.x * 128 + wave * 16;

    const size_t rowoff = (size_t)bh * SEQ;
    int qrow = qbase + (lane & 15);
    const __bf16* qrowp = Qb + (rowoff + qrow) * DH;
    v16bf aq0 = load_a16x32(qrowp, lane);
    v16bf aq1 = load_a16x32(qrowp + 32, lane);

    float q2v[8];
    #pragma unroll
    for (int r = 0; r < 8; ++r)
        q2v[r] = q2[rowoff + qbase + r + ((lane >> 4) << 3)];

    v8f acc[4] = {{}, {}, {}, {}};
    float den[8];
    #pragma unroll
    for (int r = 0; r < 8; ++r) den[r] = 0.f;

    __bf16* swp = &sw[wave][0];

    for (int j0 = 0; j0 < SEQ; j0 += 32) {
        // WGP-scope prefetch of next j-chunk's K rows and V slices
        if (j0 + 32 < SEQ) {
            __builtin_prefetch(Kb + (rowoff + j0 + 32 + (lane & 15)) * DH, 0, 3);
            __builtin_prefetch(Vt + ((size_t)bh * DH + (lane & 15)) * SEQ + j0 + 32, 0, 3);
        }
        // ---- scores for two 16x16 tiles (j0..j0+15, j0+16..j0+31) ----
        #pragma unroll
        for (int t = 0; t < 2; ++t) {
            int j = j0 + t * 16 + (lane & 15);
            const __bf16* kb = Kb + (rowoff + j) * DH + ((lane >> 4) << 4);
            v16bf b0 = *(const v16bf*)kb;            // dh 0..31 slice
            v16bf b1 = *(const v16bf*)(kb + 32);     // dh 32..63 slice
            v8f sc = {};
            sc = wmma_bf16(aq0, b0, sc);
            sc = wmma_bf16(aq1, b1, sc);
            float kj = k2[rowoff + j];
            float mj = mass[rowoff + j];
            #pragma unroll
            for (int r = 0; r < 8; ++r) {
                float d2 = q2v[r] + kj - 2.0f * sc[r];
                float dd = sqrtf(fmaxf(d2, 0.0f)) + EPSV;
                float wv = mj / (dd * dd);
                den[r] += wv;
                swp[(r + ((lane >> 4) << 3)) * 32 + t * 16 + (lane & 15)] = (__bf16)wv;
            }
        }
        // ---- reload scores in A-layout (per-wave LDS transpose) ----
        const __bf16* rowp = swp + (lane & 15) * 32;
        v16bf aw = load_a16x32(rowp, lane);
        // ---- num += W(16x32) * V(32x64) ----
        #pragma unroll
        for (int n = 0; n < 4; ++n) {
            const __bf16* vb = Vt + ((size_t)bh * DH + n * 16 + (lane & 15)) * SEQ
                                 + j0 + ((lane >> 4) << 4);
            v16bf bv = *(const v16bf*)vb;
            acc[n] = wmma_bf16(aw, bv, acc[n]);
        }
    }

    // ---- finish den: reduce across the 16 lanes of each half-wave ----
    #pragma unroll
    for (int r = 0; r < 8; ++r) {
        float d = den[r];
        d += __shfl_xor(d, 1);
        d += __shfl_xor(d, 2);
        d += __shfl_xor(d, 4);
        d += __shfl_xor(d, 8);
        den[r] = 1.0f / d;
    }

    // ---- normalize and write attention output bf16 in [B*S, D] layout ----
    #pragma unroll
    for (int n = 0; n < 4; ++n) {
        int col = h * DH + n * 16 + (lane & 15);
        #pragma unroll
        for (int r = 0; r < 8; ++r) {
            int i = qbase + r + ((lane >> 4) << 3);
            float o = acc[n][r] * den[r];
            AO[((size_t)(b_ * SEQ + i)) * DM + col] = (__bf16)o;
        }
    }
}

// ---------------- host side ----------------
extern "C" void kernel_launch(void* const* d_in, const int* in_sizes, int n_in,
                              void* d_out, int out_size, void* d_ws, size_t ws_size,
                              hipStream_t stream) {
    const float* x  = (const float*)d_in[0];
    const float* Wq = (const float*)d_in[1];
    const float* bq = (const float*)d_in[2];
    const float* Wk = (const float*)d_in[3];
    const float* bk = (const float*)d_in[4];
    const float* Wv = (const float*)d_in[5];
    const float* bv = (const float*)d_in[6];
    const float* Wo = (const float*)d_in[7];
    const float* bo = (const float*)d_in[8];
    // d_in[9] = G : cancels under row normalization, unused.

    char* ws = (char*)d_ws;
    size_t off = 0;
    auto alloc = [&](size_t bytes) -> void* {
        size_t o = (off + 255) & ~(size_t)255;
        off = o + bytes;
        return (void*)(ws + o);
    };

    const size_t XE = (size_t)ROWS * DM;     // 4096*1024 elements
    const size_t WE = (size_t)DM * DM;       // 1024*1024 elements
    __bf16* xb  = (__bf16*)alloc(XE * 2);
    __bf16* wqb = (__bf16*)alloc(WE * 2);
    __bf16* wkb = (__bf16*)alloc(WE * 2);
    __bf16* wvb = (__bf16*)alloc(WE * 2);
    __bf16* wob = (__bf16*)alloc(WE * 2);
    __bf16* Qb  = (__bf16*)alloc(XE * 2);    // [B,H,S,Dh]
    __bf16* Kb  = (__bf16*)alloc(XE * 2);    // [B,H,S,Dh]
    __bf16* Vt  = (__bf16*)alloc(XE * 2);    // [B,H,Dh,S]
    __bf16* AO  = (__bf16*)alloc(XE * 2);    // [B*S, D]
    float*  q2  = (float*)alloc((size_t)BH * SEQ * 4);
    float*  k2  = (float*)alloc((size_t)BH * SEQ * 4);
    float*  ms  = (float*)alloc((size_t)BH * SEQ * 4);

    // 1) casts
    cast_f32_bf16_k<<<1024, 256, 0, stream>>>(x,  xb,  (int)XE);
    cast_f32_bf16_k<<<1024, 256, 0, stream>>>(Wq, wqb, (int)WE);
    cast_f32_bf16_k<<<1024, 256, 0, stream>>>(Wk, wkb, (int)WE);
    cast_f32_bf16_k<<<1024, 256, 0, stream>>>(Wv, wvb, (int)WE);
    cast_f32_bf16_k<<<1024, 256, 0, stream>>>(Wo, wob, (int)WE);

    // 2) QKV projections (32x64 output tile per wave)
    const int gemmBlocks = (ROWS / 32) * (DM / 64) * 32 / 256;  // 256
    gemm_bf16_k<<<gemmBlocks, 256, 0, stream>>>(xb, wqb, bq, 0, Qb, nullptr);
    gemm_bf16_k<<<gemmBlocks, 256, 0, stream>>>(xb, wkb, bk, 0, Kb, nullptr);
    gemm_bf16_k<<<gemmBlocks, 256, 0, stream>>>(xb, wvb, bv, 1, Vt, nullptr);

    // 3) row norms
    norms_k<<<(BH * SEQ) / 256, 256, 0, stream>>>(Qb, Kb, q2, k2, ms);

    // 4) streaming attention
    dim3 agrid(SEQ / 128, BH);
    attn_k<<<agrid, 256, 0, stream>>>(Qb, Kb, Vt, q2, k2, ms, AO);

    // 5) output projection -> f32 d_out
    gemm_bf16_k<<<gemmBlocks, 256, 0, stream>>>(AO, wob, bo, 2, nullptr, (float*)d_out);
}
